// CRF_72980084293961
// MI455X (gfx1250) — matrix operand order repaired and checked by reading, not compiled, not run on hardware
//
#include <hip/hip_runtime.h>
#include <hip/hip_bf16.h>

#define T_LEN 512
#define B_DIM 256
#define N_DIM 128

#define LOG2E 1.4426950408889634f
#define LN2   0.6931471805599453f

typedef __attribute__((ext_vector_type(16))) _Float16 v16h;
typedef __attribute__((ext_vector_type(8)))  _Float16 v8h;
typedef __attribute__((ext_vector_type(8)))  float    v8f;

union U16h { v16h v; v8h h[2]; };

// Order-preserving float -> uint key (monotonic): LDS atomicMax on uint
// implements float max deterministically.
__device__ __forceinline__ unsigned fkey(float f) {
    unsigned u = __float_as_uint(f);
    return (u & 0x80000000u) ? ~u : (u | 0x80000000u);
}
__device__ __forceinline__ float fdec(unsigned u) {
    return (u & 0x80000000u) ? __uint_as_float(u & 0x7fffffffu)
                             : __uint_as_float(~u);
}

// Max across each 16-lane row via DPP16 row_xmask (pure VALU, no LDS pipe).
__device__ __forceinline__ float rowmax16(float x) {
    float y;
    y = __uint_as_float(__builtin_amdgcn_update_dpp(
            __float_as_uint(x), __float_as_uint(x), 0x161, 0xf, 0xf, false));
    x = fmaxf(x, y);
    y = __uint_as_float(__builtin_amdgcn_update_dpp(
            __float_as_uint(x), __float_as_uint(x), 0x162, 0xf, 0xf, false));
    x = fmaxf(x, y);
    y = __uint_as_float(__builtin_amdgcn_update_dpp(
            __float_as_uint(x), __float_as_uint(x), 0x164, 0xf, 0xf, false));
    x = fmaxf(x, y);
    y = __uint_as_float(__builtin_amdgcn_update_dpp(
            __float_as_uint(x), __float_as_uint(x), 0x168, 0xf, 0xf, false));
    return fmaxf(x, y);
}

// Sum across each 16-lane row via DPP16 row_xmask.
__device__ __forceinline__ float rowsum16(float x) {
    float y;
    y = __uint_as_float(__builtin_amdgcn_update_dpp(
            __float_as_uint(x), __float_as_uint(x), 0x161, 0xf, 0xf, false));
    x += y;
    y = __uint_as_float(__builtin_amdgcn_update_dpp(
            __float_as_uint(x), __float_as_uint(x), 0x162, 0xf, 0xf, false));
    x += y;
    y = __uint_as_float(__builtin_amdgcn_update_dpp(
            __float_as_uint(x), __float_as_uint(x), 0x164, 0xf, 0xf, false));
    x += y;
    y = __uint_as_float(__builtin_amdgcn_update_dpp(
            __float_as_uint(x), __float_as_uint(x), 0x168, 0xf, 0xf, false));
    return x + y;
}

// One forward step in the log2 domain, single barrier.
// Reads P from PR, writes next P to PW.  EV holds emissions*log2e; refilled t+2.
// mkey rotation: atomics -> mkey[mt], read prev max from mkey[mr], zero mkey[mz].
#define STEP_BODY(tcur, EV, PR, PW)                                              \
  do {                                                                           \
    /* previous step's row max (atomics completed before last barrier) */        \
    float mnew[8];                                                               \
    _Pragma("unroll")                                                            \
    for (int v = 0; v < 8; ++v) mnew[v] = fdec(mkey[mr][v + 8 * hi]);            \
    U16h a0, a1, a2, a3;                                                         \
    const _Float16* rp = &PR[li][0];                                             \
    const int o = 8 * hi;                                                        \
    a0.h[0] = *(const v8h*)(rp + o);       a0.h[1] = *(const v8h*)(rp + o + 16); \
    a1.h[0] = *(const v8h*)(rp + 32 + o);  a1.h[1] = *(const v8h*)(rp + 48 + o); \
    a2.h[0] = *(const v8h*)(rp + 64 + o);  a2.h[1] = *(const v8h*)(rp + 80 + o); \
    a3.h[0] = *(const v8h*)(rp + 96 + o);  a3.h[1] = *(const v8h*)(rp + 112 + o);\
    /* two independent WMMA chains -> overlap LDS latency across chunks */       \
    v8f acc0 = {}, acc1 = {};                                                    \
    acc0 = __builtin_amdgcn_wmma_f32_16x16x32_f16(false, a0.v, false,            \
               bfrag[0].v, (short)0, acc0, false, false);                        \
    acc1 = __builtin_amdgcn_wmma_f32_16x16x32_f16(false, a1.v, false,            \
               bfrag[1].v, (short)0, acc1, false, false);                        \
    acc0 = __builtin_amdgcn_wmma_f32_16x16x32_f16(false, a2.v, false,            \
               bfrag[2].v, (short)0, acc0, false, false);                        \
    acc1 = __builtin_amdgcn_wmma_f32_16x16x32_f16(false, a3.v, false,            \
               bfrag[3].v, (short)0, acc1, false, false);                        \
    v8f acc = acc0 + acc1;                                                       \
    _Pragma("unroll")                                                            \
    for (int v = 0; v < 8; ++v) {                                                \
        float sc = mrow[v] + __builtin_amdgcn_logf(acc[v]) + EV[v];              \
        alpha[v] = ((tcur) < lenr[v]) ? sc : alpha[v];                           \
    }                                                                            \
    if ((tcur) + 2 < T_LEN) {                                                    \
        const float* ep2 = emit + ((size_t)((tcur) + 2) * B_DIM + b0) * N_DIM + col; \
        _Pragma("unroll")                                                        \
        for (int v = 0; v < 8; ++v) EV[v] = ep2[(v + 8 * hi) * N_DIM] * LOG2E;   \
    }                                                                            \
    /* next P with deferred shift mnew = max(alpha_{t-1}); exponent <= ~15 */    \
    _Pragma("unroll")                                                            \
    for (int v = 0; v < 8; ++v) {                                                \
        int r = v + 8 * hi;                                                      \
        float e = fminf(alpha[v] - mnew[v], 15.5f);                              \
        PW[r][col] = (_Float16)__builtin_amdgcn_exp2f(e);                        \
    }                                                                            \
    /* this step's row maxes for t+1 (off critical path) */                      \
    float tmx[8];                                                                \
    _Pragma("unroll")                                                            \
    for (int v = 0; v < 8; ++v) tmx[v] = rowmax16(alpha[v]);                     \
    if (li == 0) {                                                               \
        _Pragma("unroll")                                                        \
        for (int v = 0; v < 8; ++v)                                              \
            atomicMax(&mkey[mt][v + 8 * hi], fkey(tmx[v]));                      \
    }                                                                            \
    if (wv == 0 && lane < 16) mkey[mz][lane] = 0u;                               \
    __syncthreads();                                                             \
    _Pragma("unroll")                                                            \
    for (int v = 0; v < 8; ++v) mrow[v] = mnew[v];                               \
    { int _t = mz; mz = mr; mr = mt; mt = _t; }                                  \
  } while (0)

// One workgroup = 16 batch rows for all 512 timesteps.
// 8 waves, wave w owns output columns [16w, 16w+16).
__global__ __launch_bounds__(256)
void crf_forward(const float* __restrict__ emit,
                 const unsigned char* __restrict__ mask,
                 const float* __restrict__ trans,
                 const float* __restrict__ strans,
                 const float* __restrict__ etrans,
                 float* __restrict__ ws_logz)
{
    __shared__ __align__(32) _Float16 Pl0[16][136];  // double-buffered P
    __shared__ __align__(32) _Float16 Pl1[16][136];
    __shared__ unsigned mkey[3][16];                 // row-max keys, 3-phase
    __shared__ int   len_s[16];
    __shared__ float psum[16][8];

    const int tid  = threadIdx.x;
    const int lane = tid & 31;
    const int wv   = tid >> 5;     // 0..7: column tile
    const int hi   = lane >> 4;    // half-wave
    const int li   = lane & 15;
    const int b0   = blockIdx.x * 16;
    const int j0   = wv * 16;
    const int col  = j0 + li;

    const float st  = strans[col];
    const float etl = etrans[col] * LOG2E;

    // B fragments: exp(trans) tile (128 x 16), register-resident.
    // Dense f16 B layout: e<8 -> K=8*hi+e, e>=8 -> K=16+8*hi+(e-8), N = lane&15.
    U16h bfrag[4];
    #pragma unroll
    for (int c = 0; c < 4; ++c) {
        #pragma unroll
        for (int e = 0; e < 16; ++e) {
            int k = 32 * c + 8 * hi + (e < 8 ? e : 8 + e);
            bfrag[c].v[e] = (_Float16)__builtin_amdgcn_exp2f(trans[k * N_DIM + col] * LOG2E);
        }
    }

    // Sequence lengths (mask is prefix-true) + zero the max-key buffers.
    if (wv == 0 && lane < 16) {
        int L = 0;
        const unsigned char* mp = mask + b0 + lane;
        for (int t = 0; t < T_LEN; ++t) L += mp[(size_t)t * B_DIM];
        len_s[lane]   = L;
        mkey[0][lane] = 0u;
        mkey[1][lane] = 0u;
        mkey[2][lane] = 0u;
    }
    __syncthreads();

    int lenr[8];
    #pragma unroll
    for (int v = 0; v < 8; ++v) lenr[v] = len_s[v + 8 * hi];

    // alpha0 in log2 domain.  (C/D layout: row v+8*hi, col lane&15)
    float alpha[8];
    #pragma unroll
    for (int v = 0; v < 8; ++v) {
        int r = v + 8 * hi;
        alpha[v] = (st + emit[(size_t)(b0 + r) * N_DIM + col]) * LOG2E;
    }

    // Prologue: exact max of alpha0 -> mkey[0]; first P into Pl1.
    #pragma unroll
    for (int v = 0; v < 8; ++v) {
        float tm = rowmax16(alpha[v]);
        if (li == 0) atomicMax(&mkey[0][v + 8 * hi], fkey(tm));
    }
    __syncthreads();
    float mrow[8];
    #pragma unroll
    for (int v = 0; v < 8; ++v) {
        int r = v + 8 * hi;
        mrow[v] = fdec(mkey[0][r]);
        Pl1[r][col] = (_Float16)__builtin_amdgcn_exp2f(alpha[v] - mrow[v]);
    }
    __syncthreads();

    // Distance-2 emission prefetch pipeline (values pre-scaled by log2e).
    float evA[8], evB[8];
    {
        const float* ep = emit + ((size_t)1 * B_DIM + b0) * N_DIM + col;
        #pragma unroll
        for (int v = 0; v < 8; ++v) evA[v] = ep[(v + 8 * hi) * N_DIM] * LOG2E;
    }
    {
        const float* ep = emit + ((size_t)2 * B_DIM + b0) * N_DIM + col;
        #pragma unroll
        for (int v = 0; v < 8; ++v) evB[v] = ep[(v + 8 * hi) * N_DIM] * LOG2E;
    }

    // mkey rotation state for t=1: target 1%3=1, read 0, zero 2.
    int mt = 1, mr = 0, mz = 2;

    int t = 1;
    #pragma unroll 1
    for (; t + 1 < T_LEN; t += 2) {
        STEP_BODY(t,     evA, Pl1, Pl0);   // odd steps read Pl1
        STEP_BODY(t + 1, evB, Pl0, Pl1);   // even steps read Pl0
    }
    STEP_BODY(t, evA, Pl1, Pl0);   // t == T_LEN-1 (odd remainder)

    // Final: exact logsumexp_j(alpha + etrans) per row, sum of 16 rows.
    // mkey[mt] was zeroed during the last step.
    float v2[8];
    #pragma unroll
    for (int v = 0; v < 8; ++v) v2[v] = alpha[v] + etl;
    #pragma unroll
    for (int v = 0; v < 8; ++v) {
        float tm = rowmax16(v2[v]);
        if (li == 0) atomicMax(&mkey[mt][v + 8 * hi], fkey(tm));
    }
    __syncthreads();
    #pragma unroll
    for (int v = 0; v < 8; ++v) {
        int r = v + 8 * hi;
        float M = fdec(mkey[mt][r]);
        float s = rowsum16(__builtin_amdgcn_exp2f(v2[v] - M));
        if (li == 0) psum[r][wv] = s;   // deterministic partials (no atomics)
    }
    __syncthreads();
    if (wv == 0 && lane < 16) {
        float ssum = 0.f;
        #pragma unroll
        for (int w = 0; w < 8; ++w) ssum += psum[lane][w];
        float lz = (fdec(mkey[mt][lane]) + __builtin_amdgcn_logf(ssum)) * LN2;
        #pragma unroll
        for (int off = 1; off < 16; off <<= 1) lz += __shfl_xor(lz, off);
        if (lane == 0) ws_logz[blockIdx.x] = lz;
    }
}

// Gold-path score: one thread per batch element, per-thread partial (deterministic).
__global__ __launch_bounds__(256)
void crf_score(const float* __restrict__ emit,
               const int* __restrict__ target,
               const unsigned char* __restrict__ mask,
               const float* __restrict__ trans,
               const float* __restrict__ strans,
               const float* __restrict__ etrans,
               float* __restrict__ ws_score)
{
    int b = threadIdx.x;
    int len = 0;
    for (int t = 0; t < T_LEN; ++t) len += mask[(size_t)t * B_DIM + b];
    int prev = target[b];
    float s = strans[prev] + emit[(size_t)b * N_DIM + prev];
    for (int t = 1; t < len; ++t) {
        int c = target[(size_t)t * B_DIM + b];
        s += emit[((size_t)t * B_DIM + b) * N_DIM + c] + trans[prev * N_DIM + c];
        prev = c;
    }
    s += etrans[prev];
    ws_score[b] = s;
}

__global__ void crf_final(const float* __restrict__ ws, float* __restrict__ out)
{
    if (threadIdx.x == 0 && blockIdx.x == 0) {
        float lz = 0.f, sc = 0.f;
        for (int i = 0; i < 16; ++i)  lz += ws[i];
        for (int i = 0; i < 256; ++i) sc += ws[16 + i];
        out[0] = (lz - sc) / 256.0f;
    }
}

extern "C" void kernel_launch(void* const* d_in, const int* in_sizes, int n_in,
                              void* d_out, int out_size, void* d_ws, size_t ws_size,
                              hipStream_t stream)
{
    const float*         emit   = (const float*)d_in[0];
    const int*           target = (const int*)d_in[1];
    const unsigned char* mask   = (const unsigned char*)d_in[2];  // numpy bool = 1 byte
    const float*         trans  = (const float*)d_in[3];
    const float*         strans = (const float*)d_in[4];
    const float*         etrans = (const float*)d_in[5];
    float* ws = (float*)d_ws;   // [0..15] logZ partials, [16..271] score partials

    crf_forward<<<16, 256, 0, stream>>>(emit, mask, trans, strans, etrans, ws);
    crf_score<<<1, 256, 0, stream>>>(emit, target, mask, trans, strans, etrans, ws + 16);
    crf_final<<<1, 1, 0, stream>>>(ws, (float*)d_out);
}
